// DecoderLayer_73280732004792
// MI455X (gfx1250) — matrix operand order repaired and checked by reading, not compiled, not run on hardware
//
#include <hip/hip_runtime.h>
#include <hip/hip_bf16.h>

typedef __attribute__((ext_vector_type(16))) _Float16 v16h;
typedef __attribute__((ext_vector_type(8)))  float    v8f;
typedef unsigned int u32x4 __attribute__((ext_vector_type(4)));
typedef int          i32x8 __attribute__((ext_vector_type(8)));
typedef int          i32x4 __attribute__((ext_vector_type(4)));

static constexpr int Bb  = 4;
static constexpr int Ss  = 1024;
static constexpr int Dd  = 1024;
static constexpr int Hh  = 16;
static constexpr int FFf = 4096;
static constexpr int Mrows = Bb * Ss;   // 4096

#define WMMA_F16(a, b, c) \
  __builtin_amdgcn_wmma_f32_16x16x32_f16(false, (a), false, (b), (short)0, (c), false, false)

#ifndef USE_TDM
#if defined(__has_builtin)
#if __has_builtin(__builtin_amdgcn_tensor_load_to_lds) && __has_builtin(__builtin_amdgcn_s_wait_tensorcnt)
#define USE_TDM 1
#else
#define USE_TDM 0
#endif
#else
#define USE_TDM 0
#endif
#endif

// ---------------------------------------------------------------------------
// Fragment loaders (ISA 7.12.2 layouts, wave32)
// A (16x32, f16): lane%16 = row M; VGPR v holds K = (v<4 ? 2v : 8+2v) + (lane>=16 ? 8 : 0), K+1
// B (32x16, f16): lane%16 = col N; VGPR v holds K = (v<4 ? 2v : 8+2v) + (lane>=16 ? 8 : 0), K+1
// C (16x16, f32): VGPR r -> row = r + (lane>=16 ? 8 : 0), col = lane%16
// ---------------------------------------------------------------------------
__device__ __forceinline__ v16h frag_a(const _Float16* base, int ld, int lane) {
  const _Float16* p = base + (lane & 15) * ld;
  const int hadd = (lane & 16) ? 8 : 0;
  v16h r;
#pragma unroll
  for (int v = 0; v < 8; ++v) {
    int k = ((v < 4) ? (2 * v) : (8 + 2 * v)) + hadd;
    r[2 * v]     = p[k];
    r[2 * v + 1] = p[k + 1];
  }
  return r;
}

__device__ __forceinline__ v16h frag_b(const _Float16* base, int ld, int lane) {
  const _Float16* p = base + (lane & 15) * ld;
  const int hadd = (lane & 16) ? 8 : 0;
  v16h r;
#pragma unroll
  for (int v = 0; v < 8; ++v) {
    int k = ((v < 4) ? (2 * v) : (8 + 2 * v)) + hadd;
    r[2 * v]     = p[k];
    r[2 * v + 1] = p[k + 1];
  }
  return r;
}

#if USE_TDM
// Issue a TDM load of a [128 rows x 32 cols] f16 tile (row stride K elements in
// global) into LDS at lds_addr, padding 4 DWORDs after every 16 DWORDs so the
// LDS row stride is 40 halfs (matches the fragment readers).
__device__ __forceinline__ void tdm_issue_tile(unsigned lds_addr, const _Float16* gsrc, int K) {
  unsigned long long ga = (unsigned long long)(uintptr_t)gsrc;
  u32x4 g0;
  g0.x = 1u;                                               // count=1 (valid user D#)
  g0.y = lds_addr;                                         // LDS byte address
  g0.z = (unsigned)ga;                                     // global_addr[31:0]
  g0.w = (unsigned)((ga >> 32) & 0x01FFFFFFu) | (2u << 30); // global_addr[56:32] | type=2
  i32x8 g1;
  // data_size=1 (2B), pad_enable, pad_interval=3 (16 DWORDs), pad_amount=3 (4 DWORDs)
  g1[0] = (1 << 16) | (1 << 20) | (3 << 22) | (3 << 25);
  unsigned td0 = (unsigned)K;        // tensor_dim0 (elements per row)
  unsigned td1 = (unsigned)Mrows;    // tensor_dim1 (rows; tiles never go OOB)
  unsigned tile0 = 32u, tile1 = 128u;
  g1[1] = (int)((td0 & 0xFFFFu) << 16);                              // dim0[15:0] @ [31:16]
  g1[2] = (int)(((td0 >> 16) & 0xFFFFu) | ((td1 & 0xFFFFu) << 16));  // dim0[31:16] | dim1[15:0]
  g1[3] = (int)(((td1 >> 16) & 0xFFFFu) | (tile0 << 16));            // dim1[31:16] | tile0
  g1[4] = (int)tile1;                                                // tile1 | tile2=0
  g1[5] = K;                                                         // dim0_stride[31:0]
  g1[6] = 0;                                                         // stride high / dim1_stride
  g1[7] = 0;
  i32x4 z4 = {0, 0, 0, 0};
  i32x8 z8 = {0, 0, 0, 0, 0, 0, 0, 0};
  __builtin_amdgcn_tensor_load_to_lds(g0, g1, z4, z4, z8, 0);
}
#endif

// ---------------------------------------------------------------------------
// f32 -> f16 conversion
// ---------------------------------------------------------------------------
__global__ void f32_to_f16_kernel(const float* __restrict__ in,
                                  _Float16* __restrict__ out, int n) {
  int i = blockIdx.x * blockDim.x + threadIdx.x;
  int stride = gridDim.x * blockDim.x;
  for (; i < n; i += stride) out[i] = (_Float16)in[i];
}

// ---------------------------------------------------------------------------
// WMMA GEMM: C[M,N] = A[M,K] @ W[N,K]^T + bias, optional ReLU.
// Block tile 128x128, K-chunk 32. 256 threads = 8 waves; wave (wm,wn) owns a
// 32x64 sub-tile = 2x4 WMMA C tiles. Staging: TDM double-buffered (preferred)
// or register-double-buffered b128 copies (fallback).
// ---------------------------------------------------------------------------
__global__ __launch_bounds__(256) void linear_wmma_kernel(
    const _Float16* __restrict__ A, const _Float16* __restrict__ W,
    const float* __restrict__ bias, float* __restrict__ outF,
    _Float16* __restrict__ outH, int M, int N, int K, int relu) {
  __shared__ _Float16 As[2][128][40];   // 80B row stride: b128-aligned, conflict-free frags
  __shared__ _Float16 Ws[2][128][40];

  const int tid  = threadIdx.x;
  const int lane = tid & 31;
  const int wave = tid >> 5;
  const int wm   = wave >> 1;          // 0..3  -> 32-row group
  const int wn   = wave & 1;           // 0..1  -> 64-col group
  const int mbase = blockIdx.y * 128;
  const int nbase = blockIdx.x * 128;

  const v8f vzero = {};
  v8f acc[2][4];
#pragma unroll
  for (int i = 0; i < 2; ++i)
#pragma unroll
    for (int j = 0; j < 4; ++j) acc[i][j] = vzero;

#if USE_TDM
  const bool issuer = (wave == 0);
  const unsigned ldsA[2] = {(unsigned)(uintptr_t)&As[0][0][0],
                            (unsigned)(uintptr_t)&As[1][0][0]};
  const unsigned ldsW[2] = {(unsigned)(uintptr_t)&Ws[0][0][0],
                            (unsigned)(uintptr_t)&Ws[1][0][0]};
  if (issuer) {
    tdm_issue_tile(ldsA[0], A + (size_t)mbase * K, K);
    tdm_issue_tile(ldsW[0], W + (size_t)nbase * K, K);
    __builtin_amdgcn_s_wait_tensorcnt(0);
  }
  __syncthreads();

  for (int kc = 0; kc < K; kc += 32) {
    const int cur = (kc >> 5) & 1;
    if (issuer && (kc + 32 < K)) {   // DMA next chunk while we compute this one
      tdm_issue_tile(ldsA[cur ^ 1], A + (size_t)mbase * K + kc + 32, K);
      tdm_issue_tile(ldsW[cur ^ 1], W + (size_t)nbase * K + kc + 32, K);
    }
    v16h af[2], bf[4];
#pragma unroll
    for (int i = 0; i < 2; ++i) af[i] = frag_a(&As[cur][wm * 32 + i * 16][0], 40, lane);
#pragma unroll
    for (int j = 0; j < 4; ++j) bf[j] = frag_b(&Ws[cur][wn * 64 + j * 16][0], 40, lane);
#pragma unroll
    for (int i = 0; i < 2; ++i)
#pragma unroll
      for (int j = 0; j < 4; ++j) acc[i][j] = WMMA_F16(af[i], bf[j], acc[i][j]);
    if (issuer) __builtin_amdgcn_s_wait_tensorcnt(0);
    __syncthreads();
  }
#else
  // Register-double-buffered copy: issue next chunk's global loads before the
  // WMMA block so HBM latency hides under matrix work.
  const int lr  = tid >> 1;            // 0..127
  const int lc0 = (tid & 1) * 8;       // 0 or 8 (halfs); second uint4 at +16
  uint4 ra0, ra1, rw0, rw1;
  auto ld_chunk = [&](int kc) {
    const _Float16* pa = A + (size_t)(mbase + lr) * K + kc + lc0;
    const _Float16* pw = W + (size_t)(nbase + lr) * K + kc + lc0;
    ra0 = *(const uint4*)pa;
    ra1 = *(const uint4*)(pa + 16);
    rw0 = *(const uint4*)pw;
    rw1 = *(const uint4*)(pw + 16);
  };
  ld_chunk(0);
  for (int kc = 0; kc < K; kc += 32) {
    *(uint4*)&As[0][lr][lc0]      = ra0;
    *(uint4*)&As[0][lr][lc0 + 16] = ra1;
    *(uint4*)&Ws[0][lr][lc0]      = rw0;
    *(uint4*)&Ws[0][lr][lc0 + 16] = rw1;
    __syncthreads();
    if (kc + 32 < K) ld_chunk(kc + 32);
    v16h af[2], bf[4];
#pragma unroll
    for (int i = 0; i < 2; ++i) af[i] = frag_a(&As[0][wm * 32 + i * 16][0], 40, lane);
#pragma unroll
    for (int j = 0; j < 4; ++j) bf[j] = frag_b(&Ws[0][wn * 64 + j * 16][0], 40, lane);
#pragma unroll
    for (int i = 0; i < 2; ++i)
#pragma unroll
      for (int j = 0; j < 4; ++j) acc[i][j] = WMMA_F16(af[i], bf[j], acc[i][j]);
    __syncthreads();
  }
#endif

  // epilogue
  const int hadd = (lane & 16) ? 8 : 0;
#pragma unroll
  for (int j = 0; j < 4; ++j) {
    int col = nbase + wn * 64 + j * 16 + (lane & 15);
    float bv = bias ? bias[col] : 0.0f;
#pragma unroll
    for (int i = 0; i < 2; ++i) {
      int rowb = mbase + wm * 32 + i * 16 + hadd;
#pragma unroll
      for (int r = 0; r < 8; ++r) {
        float val = acc[i][j][r] + bv;
        if (relu) val = fmaxf(val, 0.0f);
        size_t idx = (size_t)(rowb + r) * N + col;
        if (outF) outF[idx] = val;
        if (outH) outH[idx] = (_Float16)val;
      }
    }
  }
}

// ---------------------------------------------------------------------------
// Flash-style attention: one block per (b, h, 64-query tile). Q/K/V are f16
// [B,S,D] with head h at columns h*64..h*64+63. Online softmax over 64-key
// chunks; scores via WMMA -> LDS f32; P@V accumulated in WMMA C registers.
// ---------------------------------------------------------------------------
__global__ __launch_bounds__(256) void attention_kernel(
    const _Float16* __restrict__ Q, const _Float16* __restrict__ Kg,
    const _Float16* __restrict__ V, _Float16* __restrict__ O) {
  __shared__ _Float16 Qs[64][72];     // [q][hd], 144B rows (b128-aligned)
  __shared__ _Float16 Ks[64][72];     // [key][hd]
  __shared__ _Float16 Vt[64][72];     // [hd][key]  (transposed for B frags)
  __shared__ float    Sc[64][72];     // scores f32
  __shared__ _Float16 Ps[64][72];     // exp(scores) f16
  __shared__ float mstat[64], lstat[64], rscale[64];

  const int tid  = threadIdx.x;
  const int lane = tid & 31;
  const int wave = tid >> 5;
  const int q0 = blockIdx.x * 64;
  const int h  = blockIdx.y;
  const int b  = blockIdx.z;
  const size_t base = (size_t)b * Ss * Dd + (size_t)h * 64;

  const int lrow = tid >> 2;           // 0..63
  const int lcg  = (tid & 3) * 8;      // 0,8,16,24 ; second uint4 at +32

  // load Q tile (regs first, then LDS)
  {
    const _Float16* pq = Q + base + (size_t)(q0 + lrow) * Dd + lcg;
    uint4 r0 = *(const uint4*)pq;
    uint4 r1 = *(const uint4*)(pq + 32);
    *(uint4*)&Qs[lrow][lcg]      = r0;
    *(uint4*)&Qs[lrow][lcg + 32] = r1;
  }
  if (tid < 64) { mstat[tid] = -3.0e38f; lstat[tid] = 0.0f; }

  const int qi  = wave >> 1;           // O tile row group (16 rows)
  const int nj0 = (wave & 1) * 2;      // O tile col groups {nj0, nj0+1}
  const v8f vzero = {};
  v8f o[2] = {vzero, vzero};
  const int hadd = (lane & 16) ? 8 : 0;

  for (int ch = 0; ch < Ss / 64; ++ch) {
    const int kb = ch * 64;
    // load K chunk (direct) and V chunk (transposed into Vt)
    {
      const _Float16* pk = Kg + base + (size_t)(kb + lrow) * Dd + lcg;
      const _Float16* pv = V  + base + (size_t)(kb + lrow) * Dd + lcg;
      uint4 k0 = *(const uint4*)pk;
      uint4 k1 = *(const uint4*)(pk + 32);
      uint4 v0 = *(const uint4*)pv;
      uint4 v1 = *(const uint4*)(pv + 32);
      *(uint4*)&Ks[lrow][lcg]      = k0;
      *(uint4*)&Ks[lrow][lcg + 32] = k1;
      const _Float16* t0 = (const _Float16*)&v0;
      const _Float16* t1 = (const _Float16*)&v1;
#pragma unroll
      for (int u = 0; u < 8; ++u) {
        Vt[lcg + u][lrow]      = t0[u];
        Vt[lcg + 32 + u][lrow] = t1[u];
      }
    }
    __syncthreads();

    // scores S = Q @ K^T * 0.125 : 16 tiles, 2 per wave
#pragma unroll
    for (int t2 = 0; t2 < 2; ++t2) {
      int tile = wave * 2 + t2;
      int sqi = tile >> 2, skj = tile & 3;
      v8f s = vzero;
      v16h a  = frag_a(&Qs[sqi * 16][0], 72, lane);
      v16h bb = frag_b(&Ks[skj * 16][0], 72, lane);
      s = WMMA_F16(a, bb, s);
      a  = frag_a(&Qs[sqi * 16][0] + 32, 72, lane);
      bb = frag_b(&Ks[skj * 16][0] + 32, 72, lane);
      s = WMMA_F16(a, bb, s);
      int col = skj * 16 + (lane & 15);
      int rb  = sqi * 16 + hadd;
#pragma unroll
      for (int r = 0; r < 8; ++r) Sc[rb + r][col] = s[r] * 0.125f;
    }
    __syncthreads();

    // online softmax, one thread per query row
    if (tid < 64) {
      float mold = mstat[tid], mx = mold;
      for (int j = 0; j < 64; ++j) mx = fmaxf(mx, Sc[tid][j]);
      float sum = 0.0f;
      for (int j = 0; j < 64; ++j) {
        float e = __expf(Sc[tid][j] - mx);
        Ps[tid][j] = (_Float16)e;
        sum += e;
      }
      float rs = __expf(mold - mx);
      lstat[tid] = lstat[tid] * rs + sum;
      mstat[tid] = mx;
      rscale[tid] = rs;
    }
    __syncthreads();

    // rescale running O and accumulate P @ V
    {
      int rb = qi * 16 + hadd;
#pragma unroll
      for (int t2 = 0; t2 < 2; ++t2)
#pragma unroll
        for (int r = 0; r < 8; ++r) o[t2][r] *= rscale[rb + r];

      v16h a0 = frag_a(&Ps[qi * 16][0], 72, lane);
      v16h a1 = frag_a(&Ps[qi * 16][0] + 32, 72, lane);
#pragma unroll
      for (int t2 = 0; t2 < 2; ++t2) {
        int nj = nj0 + t2;
        v16h b0 = frag_b(&Vt[nj * 16][0], 72, lane);
        o[t2] = WMMA_F16(a0, b0, o[t2]);
        b0 = frag_b(&Vt[nj * 16][0] + 32, 72, lane);
        o[t2] = WMMA_F16(a1, b0, o[t2]);
      }
    }
    __syncthreads();
  }

  // final normalization + store (f16, heads re-interleaved into [B,S,D])
  {
    int rb = qi * 16 + hadd;
#pragma unroll
    for (int t2 = 0; t2 < 2; ++t2) {
      int col = (nj0 + t2) * 16 + (lane & 15);
#pragma unroll
      for (int r = 0; r < 8; ++r) {
        int row = rb + r;
        float val = o[t2][r] / lstat[row];
        O[base + (size_t)(q0 + row) * Dd + col] = (_Float16)val;
      }
    }
  }
}

// ---------------------------------------------------------------------------
// out = LayerNorm(a + b) * gamma + beta ; one block per row of D=1024
// ---------------------------------------------------------------------------
__global__ __launch_bounds__(256) void add_ln_kernel(
    const float* __restrict__ a, const float* __restrict__ b,
    const float* __restrict__ gamma, const float* __restrict__ beta,
    float* __restrict__ outF, _Float16* __restrict__ outH) {
  __shared__ float red[256];
  __shared__ float s_mean, s_rstd;
  const int row = blockIdx.x, tid = threadIdx.x;
  const float* pa = a + (size_t)row * Dd;
  const float* pb = b + (size_t)row * Dd;

  float x[4];
  float sum = 0.0f;
#pragma unroll
  for (int i = 0; i < 4; ++i) {
    int c = tid + i * 256;
    x[i] = pa[c] + pb[c];
    sum += x[i];
  }
  red[tid] = sum;
  __syncthreads();
  for (int s = 128; s > 0; s >>= 1) {
    if (tid < s) red[tid] += red[tid + s];
    __syncthreads();
  }
  if (tid == 0) s_mean = red[0] * (1.0f / Dd);
  __syncthreads();
  const float m = s_mean;

  float vs = 0.0f;
#pragma unroll
  for (int i = 0; i < 4; ++i) {
    float d = x[i] - m;
    vs += d * d;
  }
  red[tid] = vs;
  __syncthreads();
  for (int s = 128; s > 0; s >>= 1) {
    if (tid < s) red[tid] += red[tid + s];
    __syncthreads();
  }
  if (tid == 0) s_rstd = rsqrtf(red[0] * (1.0f / Dd) + 1e-5f);
  __syncthreads();
  const float rstd = s_rstd;

#pragma unroll
  for (int i = 0; i < 4; ++i) {
    int c = tid + i * 256;
    float val = gamma[c] * (x[i] - m) * rstd + beta[c];
    size_t idx = (size_t)row * Dd + c;
    outF[idx] = val;
    if (outH) outH[idx] = (_Float16)val;
  }
}

// ---------------------------------------------------------------------------
// Host-side orchestration
// ---------------------------------------------------------------------------
extern "C" void kernel_launch(void* const* d_in, const int* in_sizes, int n_in,
                              void* d_out, int out_size, void* d_ws, size_t ws_size,
                              hipStream_t stream) {
  (void)in_sizes; (void)n_in; (void)out_size; (void)ws_size;

  const float* x   = (const float*)d_in[0];
  const float* y   = (const float*)d_in[1];
  const float* io  = (const float*)d_in[3];
  const float* cy  = (const float*)d_in[4];
  const float* iqw = (const float*)d_in[5];  const float* iqb = (const float*)d_in[6];
  const float* ikw = (const float*)d_in[7];  const float* ikb = (const float*)d_in[8];
  const float* ivw = (const float*)d_in[9];  const float* ivb = (const float*)d_in[10];
  const float* akw = (const float*)d_in[13]; const float* akb = (const float*)d_in[14];
  const float* avw = (const float*)d_in[15]; const float* avb = (const float*)d_in[16];
  const float* sow = (const float*)d_in[17]; const float* sob = (const float*)d_in[18];
  const float* cow = (const float*)d_in[19]; const float* cob = (const float*)d_in[20];
  const float* f1w = (const float*)d_in[21]; const float* f1b = (const float*)d_in[22];
  const float* f2w = (const float*)d_in[23]; const float* f2b = (const float*)d_in[24];
  const float* g1  = (const float*)d_in[25]; const float* be1 = (const float*)d_in[26];
  const float* g2  = (const float*)d_in[27]; const float* be2 = (const float*)d_in[28];
  const float* g3  = (const float*)d_in[29]; const float* be3 = (const float*)d_in[30];

  const size_t ND = (size_t)Mrows * Dd;    // 4M
  const size_t NF = (size_t)Mrows * FFf;   // 16M
  const size_t DD = (size_t)Dd * Dd;       // 1M
  const size_t FD = (size_t)FFf * Dd;      // 4M

  char* w = (char*)d_ws;
  size_t off = 0;
  auto alloc = [&](size_t bytes) -> void* {
    size_t o = (off + 255) & ~(size_t)255;
    off = o + bytes;
    return (void*)(w + o);
  };

  _Float16* io16   = (_Float16*)alloc(ND * 2);
  _Float16* x16    = (_Float16*)alloc(ND * 2);
  _Float16* cy16   = (_Float16*)alloc(ND * 2);
  _Float16* q16    = (_Float16*)alloc(ND * 2);
  _Float16* k16    = (_Float16*)alloc(ND * 2);
  _Float16* v16    = (_Float16*)alloc(ND * 2);
  _Float16* attn16 = (_Float16*)alloc(ND * 2);
  _Float16* y2_16  = (_Float16*)alloc(ND * 2);
  _Float16* h16    = (_Float16*)alloc(NF * 2);
  float*    tmp32  = (float*)alloc(ND * 4);
  float*    y1_32  = (float*)alloc(ND * 4);
  float*    y2_32  = (float*)alloc(ND * 4);
  _Float16* iqw16  = (_Float16*)alloc(DD * 2);
  _Float16* ikw16  = (_Float16*)alloc(DD * 2);
  _Float16* ivw16  = (_Float16*)alloc(DD * 2);
  _Float16* akw16  = (_Float16*)alloc(DD * 2);
  _Float16* avw16  = (_Float16*)alloc(DD * 2);
  _Float16* sow16  = (_Float16*)alloc(DD * 2);
  _Float16* cow16  = (_Float16*)alloc(DD * 2);
  _Float16* f1w16  = (_Float16*)alloc(FD * 2);
  _Float16* f2w16  = (_Float16*)alloc(FD * 2);

  auto conv = [&](const float* src, _Float16* dst, size_t n) {
    f32_to_f16_kernel<<<2048, 256, 0, stream>>>(src, dst, (int)n);
  };
  conv(io, io16, ND);  conv(x, x16, ND);  conv(cy, cy16, ND);
  conv(iqw, iqw16, DD); conv(ikw, ikw16, DD); conv(ivw, ivw16, DD);
  conv(akw, akw16, DD); conv(avw, avw16, DD);
  conv(sow, sow16, DD); conv(cow, cow16, DD);
  conv(f1w, f1w16, FD); conv(f2w, f2w16, FD);

  auto gemm = [&](const _Float16* A, const _Float16* Wm, const float* bias,
                  float* oF, _Float16* oH, int M, int N, int K, int relu) {
    dim3 g(N / 128, M / 128);
    linear_wmma_kernel<<<g, 256, 0, stream>>>(A, Wm, bias, oF, oH, M, N, K, relu);
  };
  dim3 attn_grid(Ss / 64, Hh, Bb);

  // --- self-attention (Q,K,V from image_outputs, image weights) ---
  gemm(io16, iqw16, iqb, nullptr, q16, Mrows, Dd, Dd, 0);
  gemm(io16, ikw16, ikb, nullptr, k16, Mrows, Dd, Dd, 0);
  gemm(io16, ivw16, ivb, nullptr, v16, Mrows, Dd, Dd, 0);
  attention_kernel<<<attn_grid, 256, 0, stream>>>(q16, k16, v16, attn16);
  gemm(attn16, sow16, sob, tmp32, nullptr, Mrows, Dd, Dd, 0);
  add_ln_kernel<<<Mrows, 256, 0, stream>>>(tmp32, y, g1, be1, y1_32, nullptr);

  // --- cross-attention (Q from constant_y w/ image_query, K,V from x w/ audio) ---
  gemm(cy16, iqw16, iqb, nullptr, q16, Mrows, Dd, Dd, 0);
  gemm(x16,  akw16, akb, nullptr, k16, Mrows, Dd, Dd, 0);
  gemm(x16,  avw16, avb, nullptr, v16, Mrows, Dd, Dd, 0);
  attention_kernel<<<attn_grid, 256, 0, stream>>>(q16, k16, v16, attn16);
  gemm(attn16, cow16, cob, tmp32, nullptr, Mrows, Dd, Dd, 0);
  add_ln_kernel<<<Mrows, 256, 0, stream>>>(tmp32, y1_32, g2, be2, y2_32, y2_16);

  // --- FFN ---
  gemm(y2_16, f1w16, f1b, nullptr, h16, Mrows, FFf, Dd, 1);   // relu fused
  gemm(h16,   f2w16, f2b, tmp32, nullptr, Mrows, Dd, FFf, 0);
  add_ln_kernel<<<Mrows, 256, 0, stream>>>(tmp32, y2_32, g3, be3, (float*)d_out, nullptr);
}
